// MyGRU_53386443489423
// MI455X (gfx1250) — compile-verified, matrix operand order checked by reading, not run on hardware
//
#include <hip/hip_runtime.h>
#include <stdint.h>

#define DEVINL __device__ __forceinline__

constexpr int VOCABN = 512;
constexpr int HIDN   = 1024;
constexpr int BATCHN = 64;
constexpr int SEQN   = 512;

typedef __attribute__((ext_vector_type(16))) __bf16          v16bf;
typedef __attribute__((ext_vector_type(16))) unsigned short  v16us;
typedef __attribute__((ext_vector_type(8)))  float           v8f;

// ---------- bf16 helpers ----------
DEVINL unsigned short f2bf(float f) {
    unsigned u = __builtin_bit_cast(unsigned, f);
    u += 0x7FFFu + ((u >> 16) & 1u);          // round-to-nearest-even
    return (unsigned short)(u >> 16);
}

// A fragment: 16x32 bf16, row-major source (lda in elements).
// lane&15 = row; lanes>=16 take K+8; halves 0..7 -> K+0..7, 8..15 -> K+16..23.
DEVINL v16bf load_a(const unsigned short* __restrict__ A, int lda, int row0, int k0, int lane) {
    const unsigned short* p = A + (size_t)(row0 + (lane & 15)) * lda + k0 + ((lane >> 4) << 3);
    v16us u;
#pragma unroll
    for (int i = 0; i < 8; ++i) { u[i] = p[i]; u[8 + i] = p[16 + i]; }
    return __builtin_bit_cast(v16bf, u);
}

// B fragment from pre-packed buffer: per (ntile, ktile) tile the 16 values each
// lane needs are contiguous (32B per lane, 512 elements per tile, ktiles inner).
DEVINL v16bf load_b_packed(const unsigned short* __restrict__ Bp, int ktiles, int nt, int kt, int lane) {
    const v16us* p = (const v16us*)(Bp + ((((size_t)nt * ktiles + kt) * 32) + lane) * 16);
    return __builtin_bit_cast(v16bf, *p);
}

DEVINL v8f wmma_bf16(v16bf a, v16bf b, v8f c) {
    return __builtin_amdgcn_wmma_f32_16x16x32_bf16(false, a, false, b, (short)0, c, false, false);
}

// ---------- conversion / packing / init ----------
__global__ void cvt_f32_bf16(const float* __restrict__ src, unsigned short* __restrict__ dst, int n) {
    int i = blockIdx.x * blockDim.x + threadIdx.x;
    if (i < n) dst[i] = f2bf(src[i]);
}

// Pack a K-major f32 weight [K,N] into WMMA B-fragment order (bf16).
__global__ void pack_b_kernel(const float* __restrict__ W, unsigned short* __restrict__ dst,
                              int K, int N) {
    int idx = blockIdx.x * blockDim.x + threadIdx.x;
    if (idx >= K * N) return;
    int ktiles = K >> 5;
    int i    = idx & 15;
    int lane = (idx >> 4) & 31;
    int tile = idx >> 9;                       // nt*ktiles + kt
    int nt   = tile / ktiles;
    int kt   = tile - nt * ktiles;
    int col  = (nt << 4) + (lane & 15);
    int kb   = (kt << 5) + ((lane >> 4) << 3);
    int k    = kb + ((i < 8) ? i : (8 + i));   // halves 8..15 -> K+16..23
    dst[idx] = f2bf(W[(size_t)k * N + col]);
}

// state [64,1024] -> Hbf row-major bf16 + Hf in 16x16 fragment layout (tiles over [4x64]).
DEVINL size_t frag_off(int row, int col) {
    int mtile = row >> 4, ntile = col >> 4;
    int lane  = (col & 15) | (((row >> 3) & 1) << 4);
    int v     = row & 7;
    return (((size_t)(mtile << 6) + ntile) * 32 + lane) * 8 + v;
}

__global__ void initH_kernel(const float* __restrict__ state, float* __restrict__ Hf,
                             unsigned short* __restrict__ Hbf) {
    int i = blockIdx.x * blockDim.x + threadIdx.x;
    if (i >= BATCHN * HIDN) return;
    float v = state[i];
    Hbf[i] = f2bf(v);
    Hf[frag_off(i >> 10, i & 1023)] = v;
}

__global__ void copyH_kernel(const float* __restrict__ Hf, float* __restrict__ dst) {
    int i = blockIdx.x * blockDim.x + threadIdx.x;
    if (i >= BATCHN * HIDN) return;
    dst[i] = Hf[frag_off(i >> 10, i & 1023)];
}

// ---------- Phase 1: xg frag[t, tile] = X[:, t, :] @ Wx + bias ----------
// One wave per (t, 16x16 tile of [B=64, H=1024]); A rows strided by SEQ*VOCAB.
__global__ void xproj_kernel(const unsigned short* __restrict__ Xbf,   // [B,T,V] bf16
                             const unsigned short* __restrict__ WxP,   // packed, ktiles=16
                             const float* __restrict__ bias,           // [H]
                             float* __restrict__ outp) {               // frag [T,256,32,8]
    int wv   = (blockIdx.x * blockDim.x + threadIdx.x) >> 5;
    int lane = threadIdx.x & 31;
    int t    = wv >> 8;
    int tile = wv & 255;
    int mtile = tile >> 6, ntile = tile & 63;
    int row0 = mtile << 4, col0 = ntile << 4;

    float bv = bias[col0 + (lane & 15)];
    v8f acc;
#pragma unroll
    for (int v = 0; v < 8; ++v) acc[v] = bv;

    const unsigned short* Abase = Xbf + (size_t)t * VOCABN;
    for (int kt = 0; kt < 16; ++kt) {
        if (kt + 4 < 16)
            __builtin_prefetch(WxP + (((size_t)ntile * 16) + kt + 4) * 512, 0, 0);
        acc = wmma_bf16(load_a(Abase, SEQN * VOCABN, row0, kt * 32, lane),
                        load_b_packed(WxP, 16, ntile, kt, lane), acc);
    }
    *(v8f*)(outp + (((size_t)t * 256 + tile) * 32 + lane) * 8) = acc;
}

// ---------- Phase 2a: r,z gates. waves 0..255 -> r (store r*H bf16), 256..511 -> z ----------
__global__ void gates_kernel(const unsigned short* __restrict__ Hbf,   // [64,1024] bf16
                             const unsigned short* __restrict__ WhrP,  // packed, ktiles=32
                             const unsigned short* __restrict__ WhzP,
                             const float* __restrict__ xrp_t,          // frag [256,32,8]
                             const float* __restrict__ xzp_t,
                             const float* __restrict__ Hf,             // frag
                             float* __restrict__ Zf,                   // frag
                             unsigned short* __restrict__ RHbf) {      // [64,1024] bf16
    int wv   = (blockIdx.x * blockDim.x + threadIdx.x) >> 5;
    int lane = threadIdx.x & 31;
    int gate = wv >> 8;
    int tile = wv & 255;
    int mtile = tile >> 6, ntile = tile & 63;
    int row0 = mtile << 4, col0 = ntile << 4;

    const float*          xg = gate ? xzp_t : xrp_t;
    const unsigned short* W  = gate ? WhzP  : WhrP;

    v8f acc = *(const v8f*)(xg + ((size_t)tile * 32 + lane) * 8);
    for (int kt = 0; kt < 32; ++kt) {
        if (kt + 4 < 32)
            __builtin_prefetch(W + (((size_t)ntile * 32) + kt + 4) * 512, 0, 0);
        acc = wmma_bf16(load_a(Hbf, HIDN, row0, kt * 32, lane),
                        load_b_packed(W, 32, ntile, kt, lane), acc);
    }

    int col   = col0 + (lane & 15);
    int rbase = row0 + ((lane >> 4) << 3);
    if (gate == 0) {
        v8f h = *(const v8f*)(Hf + ((size_t)tile * 32 + lane) * 8);
#pragma unroll
        for (int v = 0; v < 8; ++v) {
            float s = 1.0f / (1.0f + __expf(-acc[v]));
            RHbf[(size_t)(rbase + v) * HIDN + col] = f2bf(s * h[v]);
        }
    } else {
        v8f z;
#pragma unroll
        for (int v = 0; v < 8; ++v) z[v] = 1.0f / (1.0f + __expf(-acc[v]));
        *(v8f*)(Zf + ((size_t)tile * 32 + lane) * 8) = z;
    }
}

// ---------- Phase 2b: hbar = tanh(xh + (r*H)@Whh); H = H*z + hbar*(1-z) ----------
__global__ void update_kernel(const unsigned short* __restrict__ RHbf,
                              const unsigned short* __restrict__ WhhP, // packed, ktiles=32
                              const float* __restrict__ xhp_t,         // frag
                              const float* __restrict__ Zf,            // frag
                              float* __restrict__ Hf,                  // frag (in/out)
                              unsigned short* __restrict__ Hbf) {      // [64,1024] bf16
    int wv   = (blockIdx.x * blockDim.x + threadIdx.x) >> 5;
    int lane = threadIdx.x & 31;
    int tile = wv & 255;
    int mtile = tile >> 6, ntile = tile & 63;
    int row0 = mtile << 4, col0 = ntile << 4;

    v8f acc = *(const v8f*)(xhp_t + ((size_t)tile * 32 + lane) * 8);
    for (int kt = 0; kt < 32; ++kt) {
        if (kt + 4 < 32)
            __builtin_prefetch(WhhP + (((size_t)ntile * 32) + kt + 4) * 512, 0, 0);
        acc = wmma_bf16(load_a(RHbf, HIDN, row0, kt * 32, lane),
                        load_b_packed(WhhP, 32, ntile, kt, lane), acc);
    }

    size_t fo = ((size_t)tile * 32 + lane) * 8;
    v8f h = *(const v8f*)(Hf + fo);
    v8f z = *(const v8f*)(Zf + fo);
    v8f hn;
#pragma unroll
    for (int v = 0; v < 8; ++v)
        hn[v] = h[v] * z[v] + tanhf(acc[v]) * (1.0f - z[v]);
    *(v8f*)(Hf + fo) = hn;

    int col   = col0 + (lane & 15);
    int rbase = row0 + ((lane >> 4) << 3);
#pragma unroll
    for (int v = 0; v < 8; ++v)
        Hbf[(size_t)(rbase + v) * HIDN + col] = f2bf(hn[v]);
}

// ---------- Phase 2c: o_t = H_new @ Who + b_o ----------
__global__ void out_kernel(const unsigned short* __restrict__ Hbf,     // [64,1024] bf16
                           const unsigned short* __restrict__ WhoP,    // packed, ktiles=32, ntiles=32
                           const float* __restrict__ bo,               // [512]
                           float* __restrict__ out_t) {                // [64,512] f32
    int wv   = (blockIdx.x * blockDim.x + threadIdx.x) >> 5;
    int lane = threadIdx.x & 31;
    int mtile = wv >> 5;
    int ntile = wv & 31;
    int row0 = mtile << 4, col0 = ntile << 4;

    float bv = bo[col0 + (lane & 15)];
    v8f acc;
#pragma unroll
    for (int v = 0; v < 8; ++v) acc[v] = bv;

    for (int kt = 0; kt < 32; ++kt) {
        if (kt + 4 < 32)
            __builtin_prefetch(WhoP + (((size_t)ntile * 32) + kt + 4) * 512, 0, 0);
        acc = wmma_bf16(load_a(Hbf, HIDN, row0, kt * 32, lane),
                        load_b_packed(WhoP, 32, ntile, kt, lane), acc);
    }

    int col   = col0 + (lane & 15);
    int rbase = row0 + ((lane >> 4) << 3);
#pragma unroll
    for (int v = 0; v < 8; ++v)
        out_t[(size_t)(rbase + v) * VOCABN + col] = acc[v];
}

// ---------- host ----------
extern "C" void kernel_launch(void* const* d_in, const int* in_sizes, int n_in,
                              void* d_out, int out_size, void* d_ws, size_t ws_size,
                              hipStream_t stream) {
    (void)in_sizes; (void)n_in; (void)out_size; (void)ws_size;

    const float* X     = (const float*)d_in[0];
    const float* state = (const float*)d_in[1];
    const float* w_xr  = (const float*)d_in[2];
    const float* w_hr  = (const float*)d_in[3];
    const float* b_r   = (const float*)d_in[4];
    const float* w_xz  = (const float*)d_in[5];
    const float* w_hz  = (const float*)d_in[6];
    const float* b_z   = (const float*)d_in[7];
    const float* w_xh  = (const float*)d_in[8];
    const float* w_hh  = (const float*)d_in[9];
    const float* b_h   = (const float*)d_in[10];
    const float* w_ho  = (const float*)d_in[11];
    const float* b_o   = (const float*)d_in[12];
    float* out = (float*)d_out;

    char* ws = (char*)d_ws;
    size_t off = 0;
    auto alloc = [&](size_t bytes) -> char* {
        char* p = ws + off;
        off += (bytes + 255) & ~(size_t)255;
        return p;
    };

    unsigned short* Xbf  = (unsigned short*)alloc((size_t)BATCHN * SEQN * VOCABN * 2);
    unsigned short* WxrP = (unsigned short*)alloc((size_t)VOCABN * HIDN * 2);
    unsigned short* WxzP = (unsigned short*)alloc((size_t)VOCABN * HIDN * 2);
    unsigned short* WxhP = (unsigned short*)alloc((size_t)VOCABN * HIDN * 2);
    unsigned short* WhrP = (unsigned short*)alloc((size_t)HIDN * HIDN * 2);
    unsigned short* WhzP = (unsigned short*)alloc((size_t)HIDN * HIDN * 2);
    unsigned short* WhhP = (unsigned short*)alloc((size_t)HIDN * HIDN * 2);
    unsigned short* WhoP = (unsigned short*)alloc((size_t)HIDN * VOCABN * 2);
    float* xrp = (float*)alloc((size_t)SEQN * BATCHN * HIDN * 4);   // frag layout
    float* xzp = (float*)alloc((size_t)SEQN * BATCHN * HIDN * 4);
    float* xhp = (float*)alloc((size_t)SEQN * BATCHN * HIDN * 4);
    float* Hf  = (float*)alloc((size_t)BATCHN * HIDN * 4);          // frag layout
    float* Zf  = (float*)alloc((size_t)BATCHN * HIDN * 4);          // frag layout
    unsigned short* Hbf  = (unsigned short*)alloc((size_t)BATCHN * HIDN * 2);
    unsigned short* RHbf = (unsigned short*)alloc((size_t)BATCHN * HIDN * 2);

    cvt_f32_bf16<<<(BATCHN * SEQN * VOCABN + 255) / 256, 256, 0, stream>>>(
        X, Xbf, BATCHN * SEQN * VOCABN);

    auto pack = [&](const float* s, unsigned short* d, int K, int N) {
        pack_b_kernel<<<(K * N + 255) / 256, 256, 0, stream>>>(s, d, K, N);
    };
    pack(w_xr, WxrP, VOCABN, HIDN);
    pack(w_xz, WxzP, VOCABN, HIDN);
    pack(w_xh, WxhP, VOCABN, HIDN);
    pack(w_hr, WhrP, HIDN, HIDN);
    pack(w_hz, WhzP, HIDN, HIDN);
    pack(w_hh, WhhP, HIDN, HIDN);
    pack(w_ho, WhoP, HIDN, VOCABN);

    initH_kernel<<<(BATCHN * HIDN + 255) / 256, 256, 0, stream>>>(state, Hf, Hbf);

    // 512*256 tiles -> 131072 waves -> 16384 blocks of 8 waves
    xproj_kernel<<<16384, 256, 0, stream>>>(Xbf, WxrP, b_r, xrp);
    xproj_kernel<<<16384, 256, 0, stream>>>(Xbf, WxzP, b_z, xzp);
    xproj_kernel<<<16384, 256, 0, stream>>>(Xbf, WxhP, b_h, xhp);

    for (int t = 0; t < SEQN; ++t) {
        const float* xr_t = xrp + (size_t)t * BATCHN * HIDN;
        const float* xz_t = xzp + (size_t)t * BATCHN * HIDN;
        const float* xh_t = xhp + (size_t)t * BATCHN * HIDN;
        gates_kernel <<<16, 256, 0, stream>>>(Hbf, WhrP, WhzP, xr_t, xz_t, Hf, Zf, RHbf);
        update_kernel<<< 8, 256, 0, stream>>>(RHbf, WhhP, xh_t, Zf, Hf, Hbf);
        out_kernel   <<< 4, 256, 0, stream>>>(Hbf, WhoP, b_o, out + (size_t)t * BATCHN * VOCABN);
    }

    copyH_kernel<<<(BATCHN * HIDN + 255) / 256, 256, 0, stream>>>(
        Hf, out + (size_t)SEQN * BATCHN * VOCABN);
}
